// GatedGCNNet_80633716015153
// MI455X (gfx1250) — matrix operand order repaired and checked by reading, not compiled, not run on hardware
//
#include <hip/hip_runtime.h>
#include <hip/hip_bf16.h>
#include <stdint.h>

// ---------------------------------------------------------------------------
// GatedGCN (L=8, D=256) for MI455X / gfx1250.
// Node-path GEMMs (Ah,Bh,Dh,Eh) are fused into one bf16 WMMA kernel that
// reads the shared activation tile once and issues 16 WMMAs per K-step.
// Edge GEMM (Ce) uses the single-weight WMMA kernel. Glue kernels handle
// edge gating (f32 atomics), 2-pass BatchNorm, pooling and the readout MLP.
// ---------------------------------------------------------------------------

#define D_DIM    256
#define NNODES   50000
#define NEDGES   100000
#define NGRAPHS  512
#define NLAYERS  8
#define BN_EPS   1e-5f
#define GATE_EPS 1e-6f

typedef __bf16 bf16_t;
typedef __attribute__((ext_vector_type(16))) __bf16 v16bf;
typedef __attribute__((ext_vector_type(8)))  __bf16 v8bf;
typedef __attribute__((ext_vector_type(8)))  float  v8f;

__device__ __forceinline__ bf16_t f2bf(float f) {
    unsigned u = __float_as_uint(f);
    unsigned short s = (unsigned short)((u + 0x7FFFu + ((u >> 16) & 1u)) >> 16);
    return __builtin_bit_cast(bf16_t, s);
}

// A-fragment: contiguous K order (matches B).
//   elems 0..15 : K = k + {0..15}   (k already includes half*16)
__device__ __forceinline__ v16bf load_afrag(const bf16_t* __restrict__ p) {
    return *(const v16bf*)(p);
}

// ---------------------------------------------------------------------------
// Weight prep: W[l][k][n] f32 -> Wt[l][n][k] bf16 (transposed so B-fragments
// are contiguous per output column).
// ---------------------------------------------------------------------------
__global__ void convert_w_kernel(const float* __restrict__ W,
                                 bf16_t* __restrict__ Wt, int total) {
    int tid = blockIdx.x * blockDim.x + threadIdx.x;
    if (tid >= total) return;
    int m   = tid >> 16;
    int rem = tid & 65535;
    int n   = rem >> 8;
    int k   = rem & 255;
    Wt[((size_t)m << 16) + ((size_t)n << 8) + k] =
        f2bf(W[((size_t)m << 16) + ((size_t)k << 8) + n]);
}

// Embedding gather: f32 copy (residual path) + bf16 copy (WMMA path).
__global__ void embed_kernel(const int* __restrict__ idx,
                             const float* __restrict__ table,
                             float* __restrict__ outf,
                             bf16_t* __restrict__ outb, int n) {
    size_t tid = (size_t)blockIdx.x * blockDim.x + threadIdx.x;
    if (tid >= (size_t)n * D_DIM) return;
    int i = (int)(tid >> 8);
    int c = (int)(tid & 255);
    float v = table[(size_t)idx[i] * D_DIM + c];
    outf[tid] = v;
    outb[tid] = f2bf(v);
}

__global__ void fill_zero_kernel(float* __restrict__ p, size_t n) {
    size_t i = (size_t)blockIdx.x * blockDim.x + threadIdx.x;
    if (i < n) p[i] = 0.0f;
}

// ---------------------------------------------------------------------------
// Fused 4-weight WMMA GEMM (node path):
//   Out_o[M x 256] = A[M x 256](bf16) * W_o + bias_o   for o = 0..3
// Block: 256 thr (8 waves); block tile 64(M) x 128(N); wave tile 16 x 64
// per output. Each A-fragment load feeds 16 v_wmma_f32_16x16x32_bf16.
// ---------------------------------------------------------------------------
__global__ void __launch_bounds__(256)
gemm4_bf16_wmma(const bf16_t* __restrict__ A,
                const bf16_t* __restrict__ Wt0, const bf16_t* __restrict__ Wt1,
                const bf16_t* __restrict__ Wt2, const bf16_t* __restrict__ Wt3,
                const float* __restrict__ bias0, const float* __restrict__ bias1,
                const float* __restrict__ bias2, const float* __restrict__ bias3,
                float* __restrict__ Out0, float* __restrict__ Out1,
                float* __restrict__ Out2, float* __restrict__ Out3, int M) {
    const int lane = threadIdx.x & 31;
    const int wave = threadIdx.x >> 5;
    const int half = lane >> 4;
    const int l15  = lane & 15;

    const int m_base = blockIdx.x * 64 + (wave >> 1) * 16;
    const int n_base = blockIdx.y * 128 + (wave & 1) * 64;

    int arow = m_base + l15;
    if (arow >= M) arow = M - 1;          // clamp; stores predicated below
    const bf16_t* aptr = A + (size_t)arow * D_DIM;

    const bf16_t* wts[4] = {Wt0, Wt1, Wt2, Wt3};
    const float*  bss[4] = {bias0, bias1, bias2, bias3};
    float*        outs[4] = {Out0, Out1, Out2, Out3};

    // per-column row offset inside each transposed weight
    const size_t brow[4] = {
        (size_t)(n_base +  0 + l15) * D_DIM, (size_t)(n_base + 16 + l15) * D_DIM,
        (size_t)(n_base + 32 + l15) * D_DIM, (size_t)(n_base + 48 + l15) * D_DIM};

    v8f acc[4][4];
#pragma unroll
    for (int o = 0; o < 4; ++o)
#pragma unroll
        for (int j = 0; j < 4; ++j) acc[o][j] = (v8f){};

#pragma unroll
    for (int k0 = 0; k0 < D_DIM; k0 += 32) {
        const int ka = k0 + half * 16;   // A: K = ka+{0..15}
        const int kb = k0 + half * 16;   // B: lane-half holds K = kb+{0..15}
        v16bf af = load_afrag(aptr + ka);

#pragma unroll
        for (int o = 0; o < 4; ++o) {
            const bf16_t* w = wts[o];
            v16bf bf0 = *(const v16bf*)(w + brow[0] + kb);
            v16bf bf1 = *(const v16bf*)(w + brow[1] + kb);
            v16bf bf2 = *(const v16bf*)(w + brow[2] + kb);
            v16bf bf3 = *(const v16bf*)(w + brow[3] + kb);
            acc[o][0] = __builtin_amdgcn_wmma_f32_16x16x32_bf16(
                false, af, false, bf0, (short)0, acc[o][0], false, false);
            acc[o][1] = __builtin_amdgcn_wmma_f32_16x16x32_bf16(
                false, af, false, bf1, (short)0, acc[o][1], false, false);
            acc[o][2] = __builtin_amdgcn_wmma_f32_16x16x32_bf16(
                false, af, false, bf2, (short)0, acc[o][2], false, false);
            acc[o][3] = __builtin_amdgcn_wmma_f32_16x16x32_bf16(
                false, af, false, bf3, (short)0, acc[o][3], false, false);
        }
    }

    // C/D layout: VGPR r, lanes 0-15 -> M=r, lanes 16-31 -> M=8+r; N=lane&15.
#pragma unroll
    for (int o = 0; o < 4; ++o) {
        float* Out = outs[o];
        const float* bias = bss[o];
#pragma unroll
        for (int r = 0; r < 8; ++r) {
            int row = m_base + half * 8 + r;
            if (row < M) {
                size_t base = (size_t)row * D_DIM + n_base + l15;
                Out[base +  0] = acc[o][0][r] + bias[n_base +  0 + l15];
                Out[base + 16] = acc[o][1][r] + bias[n_base + 16 + l15];
                Out[base + 32] = acc[o][2][r] + bias[n_base + 32 + l15];
                Out[base + 48] = acc[o][3][r] + bias[n_base + 48 + l15];
            }
        }
    }
}

// ---------------------------------------------------------------------------
// Single-weight WMMA GEMM (edge path): Out = A * W + bias.
// ---------------------------------------------------------------------------
__global__ void __launch_bounds__(256)
gemm_bf16_wmma(const bf16_t* __restrict__ A,
               const bf16_t* __restrict__ Wt,
               const float* __restrict__ bias,
               float* __restrict__ Out, int M) {
    const int lane = threadIdx.x & 31;
    const int wave = threadIdx.x >> 5;
    const int half = lane >> 4;
    const int l15  = lane & 15;

    const int m_base = blockIdx.x * 64 + (wave >> 1) * 16;
    const int n_base = blockIdx.y * 128 + (wave & 1) * 64;

    int arow = m_base + l15;
    if (arow >= M) arow = M - 1;
    const bf16_t* aptr = A + (size_t)arow * D_DIM;

    const bf16_t* b0p = Wt + (size_t)(n_base +  0 + l15) * D_DIM;
    const bf16_t* b1p = Wt + (size_t)(n_base + 16 + l15) * D_DIM;
    const bf16_t* b2p = Wt + (size_t)(n_base + 32 + l15) * D_DIM;
    const bf16_t* b3p = Wt + (size_t)(n_base + 48 + l15) * D_DIM;

    v8f acc0 = {}, acc1 = {}, acc2 = {}, acc3 = {};

#pragma unroll
    for (int k0 = 0; k0 < D_DIM; k0 += 32) {
        const int ka = k0 + half * 16;
        const int kb = k0 + half * 16;
        v16bf af  = load_afrag(aptr + ka);
        v16bf bf0 = *(const v16bf*)(b0p + kb);
        v16bf bf1 = *(const v16bf*)(b1p + kb);
        v16bf bf2 = *(const v16bf*)(b2p + kb);
        v16bf bf3 = *(const v16bf*)(b3p + kb);
        acc0 = __builtin_amdgcn_wmma_f32_16x16x32_bf16(false, af, false, bf0,
                                                       (short)0, acc0, false, false);
        acc1 = __builtin_amdgcn_wmma_f32_16x16x32_bf16(false, af, false, bf1,
                                                       (short)0, acc1, false, false);
        acc2 = __builtin_amdgcn_wmma_f32_16x16x32_bf16(false, af, false, bf2,
                                                       (short)0, acc2, false, false);
        acc3 = __builtin_amdgcn_wmma_f32_16x16x32_bf16(false, af, false, bf3,
                                                       (short)0, acc3, false, false);
    }

#pragma unroll
    for (int r = 0; r < 8; ++r) {
        int row = m_base + half * 8 + r;
        if (row < M) {
            size_t base = (size_t)row * D_DIM + n_base + l15;
            Out[base +  0] = acc0[r] + bias[n_base +  0 + l15];
            Out[base + 16] = acc1[r] + bias[n_base + 16 + l15];
            Out[base + 32] = acc2[r] + bias[n_base + 32 + l15];
            Out[base + 48] = acc3[r] + bias[n_base + 48 + l15];
        }
    }
}

// ---------------------------------------------------------------------------
// Edge gate: g = Dh[src] + Eh[dst] + Ce (Ce overwritten in-place with g),
// sigma = sigmoid(g); atomic segment-sums into num/den (per dst node).
// ---------------------------------------------------------------------------
__global__ void edge_gate_kernel(const int* __restrict__ src,
                                 const int* __restrict__ dst,
                                 const float* __restrict__ Dh,
                                 const float* __restrict__ Eh,
                                 const float* __restrict__ Bh,
                                 float* __restrict__ Ce,
                                 float* __restrict__ num,
                                 float* __restrict__ den) {
    size_t tid = (size_t)blockIdx.x * blockDim.x + threadIdx.x;
    if (tid >= (size_t)NEDGES * D_DIM) return;
    int e = (int)(tid >> 8);
    int c = (int)(tid & 255);
    int s = src[e], d = dst[e];
    float g = Dh[(size_t)s * D_DIM + c] + Eh[(size_t)d * D_DIM + c] + Ce[tid];
    float sig = 1.0f / (1.0f + __expf(-g));
    Ce[tid] = g;
    atomicAdd(&num[(size_t)d * D_DIM + c], Bh[(size_t)s * D_DIM + c] * sig);
    atomicAdd(&den[(size_t)d * D_DIM + c], sig);
}

// ---------------------------------------------------------------------------
// Column reduction for BN (sum, sumsq). If num!=nullptr, first fuse the gated
// aggregation: X := X + num/(den+eps)  (node path, X = Ah).
// ---------------------------------------------------------------------------
__global__ void combine_reduce_kernel(float* __restrict__ X,
                                      const float* __restrict__ num,
                                      const float* __restrict__ den,
                                      float* __restrict__ sums,
                                      float* __restrict__ sumsq, int rows) {
    const int colW = threadIdx.x & 31;
    const int rl   = threadIdx.x >> 5;
    const int c    = blockIdx.x * 32 + colW;
    const int slices = gridDim.y;

    float s = 0.0f, s2 = 0.0f;
    for (int r = blockIdx.y * 8 + rl; r < rows; r += slices * 8) {
        size_t off = (size_t)r * D_DIM + c;
        float x = X[off];
        if (num) {
            x += num[off] / (den[off] + GATE_EPS);
            X[off] = x;
        }
        s += x;
        s2 += x * x;
    }
    __shared__ float sh[512];
    sh[threadIdx.x]       = s;
    sh[256 + threadIdx.x] = s2;
    __syncthreads();
    if (threadIdx.x < 32) {
        float a = 0.0f, b = 0.0f;
#pragma unroll
        for (int i = 0; i < 8; ++i) {
            a += sh[colW + i * 32];
            b += sh[256 + colW + i * 32];
        }
        atomicAdd(&sums[c], a);
        atomicAdd(&sumsq[c], b);
    }
}

// BN apply + ReLU + residual; re-materialize bf16 activation for next layer.
__global__ void bn_apply_kernel(const float* __restrict__ X,
                                const float* __restrict__ sums,
                                const float* __restrict__ sumsq,
                                const float* __restrict__ gamma,
                                const float* __restrict__ beta,
                                float* __restrict__ resF,
                                bf16_t* __restrict__ resB, int rows) {
    size_t tid = (size_t)blockIdx.x * blockDim.x + threadIdx.x;
    if (tid >= (size_t)rows * D_DIM) return;
    int c = (int)(tid & 255);
    float invN = 1.0f / (float)rows;
    float mu   = sums[c] * invN;
    float var  = sumsq[c] * invN - mu * mu;
    float rstd = rsqrtf(var + BN_EPS);
    float y = (X[tid] - mu) * rstd * gamma[c] + beta[c];
    y = fmaxf(y, 0.0f);
    float h = resF[tid] + y;
    resF[tid] = h;
    resB[tid] = f2bf(h);
}

// Graph mean-pool scatter.
__global__ void graph_pool_kernel(const float* __restrict__ hf,
                                  const int* __restrict__ gid,
                                  float* __restrict__ hg,
                                  float* __restrict__ counts) {
    size_t tid = (size_t)blockIdx.x * blockDim.x + threadIdx.x;
    if (tid >= (size_t)NNODES * D_DIM) return;
    int i = (int)(tid >> 8);
    int c = (int)(tid & 255);
    int g = gid[i];
    atomicAdd(&hg[(size_t)g * D_DIM + c], hf[tid]);
    if (c == 0) atomicAdd(&counts[g], 1.0f);
}

// MLPReadout: 256 -> 128 -> 64 -> 1.  One block (128 thr) per graph.
__global__ void __launch_bounds__(128)
mlp_kernel(const float* __restrict__ hg, const float* __restrict__ counts,
           const float* __restrict__ w0, const float* __restrict__ b0,
           const float* __restrict__ w1, const float* __restrict__ b1,
           const float* __restrict__ w2, const float* __restrict__ b2,
           float* __restrict__ out) {
    int g = blockIdx.x, t = threadIdx.x;
    __shared__ float x[256];
    __shared__ float y0[128];
    __shared__ float y1[64];
    float inv = 1.0f / fmaxf(counts[g], 1.0f);
    x[t]       = hg[(size_t)g * 256 + t] * inv;
    x[t + 128] = hg[(size_t)g * 256 + t + 128] * inv;
    __syncthreads();
    float a = b0[t];
    for (int k = 0; k < 256; ++k) a += x[k] * w0[k * 128 + t];
    y0[t] = fmaxf(a, 0.0f);
    __syncthreads();
    if (t < 64) {
        float b = b1[t];
        for (int k = 0; k < 128; ++k) b += y0[k] * w1[k * 64 + t];
        y1[t] = fmaxf(b, 0.0f);
    }
    __syncthreads();
    if (t == 0) {
        float c = b2[0];
        for (int k = 0; k < 64; ++k) c += y1[k] * w2[k];
        out[g] = c;
    }
}

// ---------------------------------------------------------------------------
extern "C" void kernel_launch(void* const* d_in, const int* in_sizes, int n_in,
                              void* d_out, int out_size, void* d_ws, size_t ws_size,
                              hipStream_t stream) {
    (void)in_sizes; (void)n_in; (void)out_size; (void)ws_size;

    const int*   h_idx = (const int*)d_in[0];
    const int*   e_idx = (const int*)d_in[1];
    const int*   src   = (const int*)d_in[2];
    const int*   dst   = (const int*)d_in[3];
    const int*   gid   = (const int*)d_in[4];
    const float* emb_h = (const float*)d_in[5];
    const float* emb_e = (const float*)d_in[6];
    const float* Aw = (const float*)d_in[7];   const float* Ab = (const float*)d_in[8];
    const float* Bw = (const float*)d_in[9];   const float* Bb = (const float*)d_in[10];
    const float* Cw = (const float*)d_in[11];  const float* Cb = (const float*)d_in[12];
    const float* Dw = (const float*)d_in[13];  const float* Db = (const float*)d_in[14];
    const float* Ew = (const float*)d_in[15];  const float* Eb = (const float*)d_in[16];
    const float* bnh_g = (const float*)d_in[17]; const float* bnh_b = (const float*)d_in[18];
    const float* bne_g = (const float*)d_in[19]; const float* bne_b = (const float*)d_in[20];
    const float* mw0 = (const float*)d_in[21]; const float* mb0 = (const float*)d_in[22];
    const float* mw1 = (const float*)d_in[23]; const float* mb1 = (const float*)d_in[24];
    const float* mw2 = (const float*)d_in[25]; const float* mb2 = (const float*)d_in[26];
    float* out = (float*)d_out;

    const size_t ND = (size_t)NNODES * D_DIM;
    const size_t ED = (size_t)NEDGES * D_DIM;
    const size_t WSZ = (size_t)NLAYERS * D_DIM * D_DIM;   // per weight tensor

    char* ws = (char*)d_ws;
    size_t off = 0;
    auto carve = [&](size_t bytes) -> char* {
        char* p = ws + off;
        off += (bytes + 255) & ~(size_t)255;
        return p;
    };
    float*  hf   = (float*)carve(ND * 4);
    float*  ef   = (float*)carve(ED * 4);
    bf16_t* hbf  = (bf16_t*)carve(ND * 2);
    bf16_t* ebf  = (bf16_t*)carve(ED * 2);
    float*  Ah   = (float*)carve(ND * 4);
    float*  Bh   = (float*)carve(ND * 4);
    float*  Dh   = (float*)carve(ND * 4);
    float*  Eh   = (float*)carve(ND * 4);
    float*  Ce   = (float*)carve(ED * 4);
    float*  numden = (float*)carve(2 * ND * 4);
    float*  num = numden;
    float*  den = numden + ND;
    bf16_t* WtA = (bf16_t*)carve(WSZ * 2);
    bf16_t* WtB = (bf16_t*)carve(WSZ * 2);
    bf16_t* WtC = (bf16_t*)carve(WSZ * 2);
    bf16_t* WtD = (bf16_t*)carve(WSZ * 2);
    bf16_t* WtE = (bf16_t*)carve(WSZ * 2);
    float*  stats = (float*)carve(4 * D_DIM * 4);
    float*  nsum = stats, *nsq = stats + 256, *esum = stats + 512, *esq = stats + 768;
    float*  hg     = (float*)carve((size_t)NGRAPHS * D_DIM * 4);
    float*  counts = (float*)carve((size_t)NGRAPHS * 4);

    // --- weight prep ---
    {
        int total = (int)WSZ;
        int blocks = (total + 255) / 256;
        convert_w_kernel<<<blocks, 256, 0, stream>>>(Aw, WtA, total);
        convert_w_kernel<<<blocks, 256, 0, stream>>>(Bw, WtB, total);
        convert_w_kernel<<<blocks, 256, 0, stream>>>(Cw, WtC, total);
        convert_w_kernel<<<blocks, 256, 0, stream>>>(Dw, WtD, total);
        convert_w_kernel<<<blocks, 256, 0, stream>>>(Ew, WtE, total);
    }

    // --- embeddings ---
    embed_kernel<<<(unsigned)((ND + 255) / 256), 256, 0, stream>>>(h_idx, emb_h, hf, hbf, NNODES);
    embed_kernel<<<(unsigned)((ED + 255) / 256), 256, 0, stream>>>(e_idx, emb_e, ef, ebf, NEDGES);

    const dim3 gemm_blk(256);
    const dim3 gemm_grid_h((NNODES + 63) / 64, 2);
    const dim3 gemm_grid_e((NEDGES + 63) / 64, 2);

    for (int l = 0; l < NLAYERS; ++l) {
        const size_t wo = (size_t)l * D_DIM * D_DIM;
        const size_t bo = (size_t)l * D_DIM;

        // fused node GEMMs (A,B,D,E) + edge GEMM (C)
        gemm4_bf16_wmma<<<gemm_grid_h, gemm_blk, 0, stream>>>(
            hbf, WtA + wo, WtB + wo, WtD + wo, WtE + wo,
            Ab + bo, Bb + bo, Db + bo, Eb + bo, Ah, Bh, Dh, Eh, NNODES);
        gemm_bf16_wmma<<<gemm_grid_e, gemm_blk, 0, stream>>>(
            ebf, WtC + wo, Cb + bo, Ce, NEDGES);

        // zero segment accumulators + BN stats
        fill_zero_kernel<<<(unsigned)((2 * ND + 255) / 256), 256, 0, stream>>>(numden, 2 * ND);
        fill_zero_kernel<<<4, 256, 0, stream>>>(stats, 4 * D_DIM);

        // edge gate + segment sums
        edge_gate_kernel<<<(unsigned)((ED + 255) / 256), 256, 0, stream>>>(
            src, dst, Dh, Eh, Bh, Ce, num, den);

        // node: h_new = Ah + num/(den+eps) (in-place into Ah) + column stats
        combine_reduce_kernel<<<dim3(8, 64), 256, 0, stream>>>(Ah, num, den, nsum, nsq, NNODES);
        // edge: column stats of gate (already in Ce)
        combine_reduce_kernel<<<dim3(8, 96), 256, 0, stream>>>(Ce, nullptr, nullptr, esum, esq, NEDGES);

        // BN + ReLU + residual
        bn_apply_kernel<<<(unsigned)((ND + 255) / 256), 256, 0, stream>>>(
            Ah, nsum, nsq, bnh_g + bo, bnh_b + bo, hf, hbf, NNODES);
        bn_apply_kernel<<<(unsigned)((ED + 255) / 256), 256, 0, stream>>>(
            Ce, esum, esq, bne_g + bo, bne_b + bo, ef, ebf, NEDGES);
    }

    // --- readout ---
    fill_zero_kernel<<<(unsigned)(((size_t)NGRAPHS * D_DIM + NGRAPHS + 255) / 256), 256, 0, stream>>>(
        hg, (size_t)NGRAPHS * D_DIM + NGRAPHS);   // hg and counts are contiguous
    graph_pool_kernel<<<(unsigned)((ND + 255) / 256), 256, 0, stream>>>(hf, gid, hg, counts);
    mlp_kernel<<<NGRAPHS, 128, 0, stream>>>(hg, counts, mw0, mb0, mw1, mb1, mw2, mb2, out);
}
